// SceneStreamerDecoderLayer_2671469658513
// MI455X (gfx1250) — matrix operand order repaired and checked by reading, not compiled, not run on hardware
//
#include <hip/hip_runtime.h>
#include <stdint.h>

#define D     256
#define H     8
#define NNODE 4096
#define NEDGE 131072
#define DFF   1024
#define EPS   1e-5f

typedef __attribute__((ext_vector_type(16))) __bf16 v16bf;
typedef __attribute__((ext_vector_type(8)))  __bf16 v8bf;
typedef __attribute__((ext_vector_type(8)))  float  v8f;

__device__ __forceinline__ unsigned short f32_to_bf16(float f) {
    unsigned u = __float_as_uint(f);
    unsigned r = 0x7FFFu + ((u >> 16) & 1u);   // round-to-nearest-even
    u += r;
    return (unsigned short)(u >> 16);
}

// ---------------------------------------------------------------- converts
__global__ void cvt_bf16_kernel(const float* __restrict__ x,
                                unsigned short* __restrict__ y, int n) {
    int i = blockIdx.x * 256 + threadIdx.x;
    if (i < n) y[i] = f32_to_bf16(x[i]);
}

// W [K,Nc] row-major f32  ->  Wt [Nc,K] row-major bf16 (i.e. column-major W)
__global__ void wcvt_kernel(const float* __restrict__ w,
                            unsigned short* __restrict__ wt, int K, int Nc) {
    int i = blockIdx.x * 256 + threadIdx.x;
    if (i < K * Nc) {
        int k = i / Nc, n = i % Nc;
        wt[(size_t)n * K + k] = f32_to_bf16(w[i]);
    }
}

// ---------------------------------------------------------------- layernorm
// one 256-thread block per row of 256; output bf16 (only ever a GEMM input)
__global__ void ln_kernel(const float* __restrict__ x,
                          const float* __restrict__ g,
                          const float* __restrict__ b,
                          unsigned short* __restrict__ y) {
    __shared__ float sm[8];
    int row = blockIdx.x, t = threadIdx.x;
    float v = x[(size_t)row * D + t];
    float s = v;
    for (int o = 16; o > 0; o >>= 1) s += __shfl_xor(s, o, 32);
    if ((t & 31) == 0) sm[t >> 5] = s;
    __syncthreads();
    float mean = 0.f;
    for (int i = 0; i < 8; i++) mean += sm[i];
    mean *= (1.f / D);
    __syncthreads();
    float c = v - mean;
    float q = c * c;
    for (int o = 16; o > 0; o >>= 1) q += __shfl_xor(q, o, 32);
    if ((t & 31) == 0) sm[t >> 5] = q;
    __syncthreads();
    float var = 0.f;
    for (int i = 0; i < 8; i++) var += sm[i];
    var *= (1.f / D);
    float out = c * rsqrtf(var + EPS) * g[t] + b[t];
    y[(size_t)row * D + t] = f32_to_bf16(out);
}

// ---------------------------------------------------------------- WMMA GEMM
// C[M,Nc] = A[M,K](bf16) x W[K,Nc] + bias (+ relu) (+ residual). W given as
// Wt[Nc,K] bf16. Each wave computes a 16x64 output tile: one A-fragment is
// reused against 4 B-fragments per K-step (2.5 b128 loads per WMMA instead
// of 4). 4 waves per block cover 64 rows. Requires M%64==0, Nc%64==0
// (true for all GEMMs here), so EXEC stays all-ones as WMMA requires.
// A-fragment: lane holds row (lane&15), K-chunk [kb,kb+16),
//             kb = 16*(lane>>4)  -> one contiguous 32B load (matches B).
// B-fragment: lane holds col (lane&15), K-chunk [kb2,kb2+16), kb2 = 16*(lane>>4)
//             -> one contiguous 32B load from the transposed weights.
__global__ void gemm_wmma_kernel(const unsigned short* __restrict__ A,
                                 const unsigned short* __restrict__ Wt,
                                 const float* __restrict__ bias,
                                 const float* __restrict__ residual,
                                 float* __restrict__ outF,
                                 unsigned short* __restrict__ outB,
                                 int M, int K, int Nc, int relu) {
    const int lane  = threadIdx.x;                       // 0..31 (wave32)
    const int tileM = blockIdx.x * 4 + threadIdx.y;      // 16-row tile index
    const int tileN = blockIdx.y;                        // 64-col tile index
    const int rlo = lane & 15, hi = lane >> 4;

    const __bf16* Ab = (const __bf16*)A;
    const __bf16* Bb = (const __bf16*)Wt;
    const size_t arow = (size_t)(tileM * 16 + rlo) * K + hi * 16;
    const size_t brow = (size_t)(tileN * 64 + rlo) * K + hi * 16;   // +nb*16*K

    v8f acc0 = {}, acc1 = {}, acc2 = {}, acc3 = {};
    for (int k = 0; k < K; k += 32) {
        v16bf a = *(const v16bf*)(Ab + arow + k);
        v16bf b0 = *(const v16bf*)(Bb + brow + k);
        v16bf b1 = *(const v16bf*)(Bb + brow + (size_t)16 * K + k);
        v16bf b2 = *(const v16bf*)(Bb + brow + (size_t)32 * K + k);
        v16bf b3 = *(const v16bf*)(Bb + brow + (size_t)48 * K + k);
        acc0 = __builtin_amdgcn_wmma_f32_16x16x32_bf16(false, a, false, b0,
                                                       (short)0, acc0, false, false);
        acc1 = __builtin_amdgcn_wmma_f32_16x16x32_bf16(false, a, false, b1,
                                                       (short)0, acc1, false, false);
        acc2 = __builtin_amdgcn_wmma_f32_16x16x32_bf16(false, a, false, b2,
                                                       (short)0, acc2, false, false);
        acc3 = __builtin_amdgcn_wmma_f32_16x16x32_bf16(false, a, false, b3,
                                                       (short)0, acc3, false, false);
    }

    const int mbase = tileM * 16 + hi * 8;               // C/D: VGPR r -> M = r + 8*hi
    v8f accs[4] = {acc0, acc1, acc2, acc3};
#pragma unroll
    for (int nb = 0; nb < 4; nb++) {
        const int n  = tileN * 64 + nb * 16 + rlo;
        const float bn = bias ? bias[n] : 0.f;
#pragma unroll
        for (int r = 0; r < 8; r++) {
            int m = mbase + r;
            float v = accs[nb][r] + bn;
            if (relu) v = v > 0.f ? v : 0.f;
            if (residual) v += residual[(size_t)m * Nc + n];
            if (outF) outF[(size_t)m * Nc + n] = v;
            if (outB) outB[(size_t)m * Nc + n] = f32_to_bf16(v);
        }
    }
}

// ---------------------------------------------------------------- attention
// ordered-float map for unsigned atomicMax
__device__ __forceinline__ unsigned fmap(float f) {
    unsigned u = __float_as_uint(f);
    return (u & 0x80000000u) ? ~u : (u | 0x80000000u);
}
__device__ __forceinline__ float funmap(unsigned u) {
    u = (u & 0x80000000u) ? (u & 0x7FFFFFFFu) : ~u;
    return __uint_as_float(u);
}

// one block per edge; thread t -> head t>>5, dim t&31
__global__ void scores_kernel(const long long* __restrict__ ei,
                              const float* __restrict__ Q,
                              const float* __restrict__ Qr,
                              const float* __restrict__ Kb,
                              const float* __restrict__ efk,
                              float* __restrict__ sc,
                              unsigned* __restrict__ segmax) {
    int e = blockIdx.x, t = threadIdx.x;
    int src = (int)ei[e], dst = (int)ei[NEDGE + e];
    float p = Q[(size_t)dst * D + t] * Kb[(size_t)src * D + t] +
              Qr[(size_t)dst * D + t] * efk[(size_t)e * D + t];
    for (int o = 16; o > 0; o >>= 1) p += __shfl_xor(p, o, 32);
    if ((t & 31) == 0) {
        float s = p * 0.17677669529663687f;   // 1/sqrt(32)
        int h = t >> 5;
        sc[(size_t)e * H + h] = s;
        atomicMax(&segmax[dst * H + h], fmap(s));
    }
}

__global__ void expsum_kernel(const long long* __restrict__ ei,
                              float* __restrict__ sc,
                              const unsigned* __restrict__ segmax,
                              float* __restrict__ segsum) {
    int i = blockIdx.x * 256 + threadIdx.x;
    if (i >= NEDGE * H) return;
    int e = i / H, h = i % H;
    int dst = (int)ei[NEDGE + e];
    float es = expf(sc[i] - funmap(segmax[dst * H + h]));
    sc[i] = es;
    atomicAdd(&segsum[dst * H + h], es);
}

__global__ void message_kernel(const long long* __restrict__ ei,
                               const float* __restrict__ sc,
                               const float* __restrict__ segsum,
                               const float* __restrict__ V,
                               const float* __restrict__ efv,
                               float* __restrict__ msg) {
    int e = blockIdx.x, t = threadIdx.x;
    int src = (int)ei[e], dst = (int)ei[NEDGE + e];
    int h = t >> 5;
    float w = sc[(size_t)e * H + h] / segsum[dst * H + h];
    atomicAdd(&msg[(size_t)dst * D + t],
              (V[(size_t)src * D + t] + efv[(size_t)e * D + t]) * w);
}

// ---------------------------------------------------------------- driver
extern "C" void kernel_launch(void* const* d_in, const int* in_sizes, int n_in,
                              void* d_out, int out_size, void* d_ws, size_t ws_size,
                              hipStream_t stream) {
    (void)in_sizes; (void)n_in; (void)out_size; (void)ws_size;
    const float*     map_token = (const float*)d_in[0];
    const float*     all_token = (const float*)d_in[1];
    const long long* a2m_ei    = (const long long*)d_in[2];
    const float*     a2m_ef    = (const float*)d_in[3];
    const long long* a2a_ei    = (const long long*)d_in[4];
    const float*     a2a_ef    = (const float*)d_in[5];
    auto P = [&](int i) { return (const float*)d_in[i]; };
    // params flat order: a2m_attn (to_q,to_k,to_v,to_q_relation,to_k_r,to_v_r,out)x(w,b)=6..19
    //                    a2a_attn same = 20..33; norms 34..43; fc1 44/45; fc2 46/47

    char* ws = (char*)d_ws;
    size_t off = 0;
    auto alloc = [&](size_t bytes) -> void* {
        void* p = ws + off;
        off += (bytes + 255) & ~(size_t)255;
        return p;
    };
    unsigned short* xn    = (unsigned short*)alloc((size_t)NNODE * D * 2);
    unsigned short* ktokb = (unsigned short*)alloc((size_t)NNODE * D * 2);
    unsigned short* relb  = (unsigned short*)alloc((size_t)NEDGE * D * 2);
    unsigned short* msgb  = (unsigned short*)alloc((size_t)NNODE * D * 2);
    unsigned short* hb    = (unsigned short*)alloc((size_t)NNODE * DFF * 2);
    unsigned short* wt    = (unsigned short*)alloc((size_t)D * DFF * 2);
    float*    Qb   = (float*)alloc((size_t)NNODE * D * 4);
    float*    Qrb  = (float*)alloc((size_t)NNODE * D * 4);
    float*    Kbuf = (float*)alloc((size_t)NNODE * D * 4);
    float*    Vbuf = (float*)alloc((size_t)NNODE * D * 4);
    float*    efk  = (float*)alloc((size_t)NEDGE * D * 4);
    float*    efv  = (float*)alloc((size_t)NEDGE * D * 4);
    float*    sc   = (float*)alloc((size_t)NEDGE * H * 4);
    unsigned* smax = (unsigned*)alloc((size_t)NNODE * H * 4);
    float*    ssum = (float*)alloc((size_t)NNODE * H * 4);
    float*    msg  = (float*)alloc((size_t)NNODE * D * 4);
    float*    x1   = (float*)alloc((size_t)NNODE * D * 4);

    auto gemm = [&](const unsigned short* A, int wi, int bi, int M, int K, int Nc,
                    const float* residual, float* outF, unsigned short* outB, int relu) {
        int tot = K * Nc;
        wcvt_kernel<<<(tot + 255) / 256, 256, 0, stream>>>(P(wi), wt, K, Nc);
        dim3 grid(M / 64, Nc / 64), blk(32, 4);
        gemm_wmma_kernel<<<grid, blk, 0, stream>>>(A, wt, P(bi), residual,
                                                   outF, outB, M, K, Nc, relu);
    };

    auto run_mha = [&](int base, const unsigned short* qtokb, const unsigned short* ktb,
                       const long long* ei, float* xout) {
        gemm(qtokb, base + 0,  base + 1,  NNODE, D, D, nullptr, Qb,   nullptr, 0); // to_q
        gemm(ktb,   base + 2,  base + 3,  NNODE, D, D, nullptr, Kbuf, nullptr, 0); // to_k
        gemm(ktb,   base + 4,  base + 5,  NNODE, D, D, nullptr, Vbuf, nullptr, 0); // to_v
        gemm(qtokb, base + 6,  base + 7,  NNODE, D, D, nullptr, Qrb,  nullptr, 0); // to_q_relation
        gemm(relb,  base + 8,  base + 9,  NEDGE, D, D, nullptr, efk,  nullptr, 0); // to_k_r
        gemm(relb,  base + 10, base + 11, NEDGE, D, D, nullptr, efv,  nullptr, 0); // to_v_r
        hipMemsetAsync(smax, 0, (size_t)NNODE * H * 4, stream);
        hipMemsetAsync(ssum, 0, (size_t)NNODE * H * 4, stream);
        hipMemsetAsync(msg,  0, (size_t)NNODE * D * 4, stream);
        scores_kernel<<<NEDGE, 256, 0, stream>>>(ei, Qb, Qrb, Kbuf, efk, sc, smax);
        expsum_kernel<<<(NEDGE * H + 255) / 256, 256, 0, stream>>>(ei, sc, smax, ssum);
        message_kernel<<<NEDGE, 256, 0, stream>>>(ei, sc, ssum, Vbuf, efv, msg);
        cvt_bf16_kernel<<<(NNODE * D + 255) / 256, 256, 0, stream>>>(msg, msgb, NNODE * D);
        gemm(msgb, base + 12, base + 13, NNODE, D, D, all_token, xout, nullptr, 0); // out proj + residual
    };

    // ---- all-to-map cross attention
    cvt_bf16_kernel<<<(NNODE * D + 255) / 256, 256, 0, stream>>>(map_token, ktokb, NNODE * D);
    ln_kernel<<<NNODE, 256, 0, stream>>>(all_token, P(34), P(35), xn);
    ln_kernel<<<NEDGE, 256, 0, stream>>>(a2m_ef, P(36), P(37), relb);
    run_mha(6, xn, ktokb, a2m_ei, x1);

    // ---- all-to-all self attention (q_tok == k_tok == LN(x1))
    ln_kernel<<<NNODE, 256, 0, stream>>>(x1, P(38), P(39), xn);
    ln_kernel<<<NEDGE, 256, 0, stream>>>(a2a_ef, P(40), P(41), relb);
    run_mha(20, xn, xn, a2a_ei, x1);   // x1 now holds x2 = input_all + attn2

    // ---- MLP: prenorm, fc1+ReLU (bf16 hidden), fc2 + residual -> d_out
    ln_kernel<<<NNODE, 256, 0, stream>>>(x1, P(42), P(43), xn);
    gemm(xn, 44, 45, NNODE, D,   DFF, nullptr,   nullptr,        hb, 1);
    gemm(hb, 46, 47, NNODE, DFF, D,   all_token, (float*)d_out,  nullptr, 0);
}
